// MAB_56495999812255
// MI455X (gfx1250) — matrix operand order repaired and checked by reading, not compiled, not run on hardware
//
#include <hip/hip_runtime.h>
#include <hip/hip_bf16.h>

// ---------------------------------------------------------------------------
// Problem constants (from reference): B=4, L=2048, DV=512, H=8, HD=64, DFF=2048
// ---------------------------------------------------------------------------
#define DVX     512
#define NH      8
#define HDIM    64
#define SEQ     2048
#define NBATCH  4
#define DFF     2048
#define BLROWS  (NBATCH * SEQ)   // 8192 token rows

typedef __attribute__((ext_vector_type(16))) __bf16 v16bf;
typedef __attribute__((ext_vector_type(8)))  __bf16 v8bf;
typedef __attribute__((ext_vector_type(8)))  float  v8f;

// ---------------------------------------------------------------------------
// WMMA fragment loaders (CDNA5 wave32 layouts, ISA 7.12.2)
//  A (16x32 bf16): lane l holds row (l&15); element e<8 -> K = (l>>4)*8+e,
//                  e>=8 -> K = 16+(l>>4)*8+(e-8)   => two 16B loads
//  B (32x16 bf16): lane l holds col (l&15); same K mapping as A
//                  => two 16B loads
// ---------------------------------------------------------------------------
__device__ __forceinline__ v16bf load_a_frag(const __bf16* __restrict__ A, int lda,
                                             int rowBase, int k0, int lane) {
    int m  = rowBase + (lane & 15);
    int kh = (lane >> 4) << 3;
    const __bf16* p = A + (size_t)m * lda + k0 + kh;
    v8bf lo = *(const v8bf*)p;
    v8bf hi = *(const v8bf*)(p + 16);
    v16bf r;
#pragma unroll
    for (int i = 0; i < 8; ++i) { r[i] = lo[i]; r[i + 8] = hi[i]; }
    return r;
}

__device__ __forceinline__ v16bf load_b_frag(const __bf16* __restrict__ Bt, int ldb,
                                             int colBase, int k0, int lane) {
    int n  = colBase + (lane & 15);
    int kh = (lane >> 4) << 3;
    const __bf16* p = Bt + (size_t)n * ldb + k0 + kh;
    v8bf lo = *(const v8bf*)p;
    v8bf hi = *(const v8bf*)(p + 16);
    v16bf r;
#pragma unroll
    for (int i = 0; i < 8; ++i) { r[i] = lo[i]; r[i + 8] = hi[i]; }
    return r;
}

__device__ __forceinline__ v8f wmma_bf16(v16bf a, v16bf b, v8f c) {
    return __builtin_amdgcn_wmma_f32_16x16x32_bf16(false, a, false, b, (short)0, c,
                                                   false, false);
}

// Per-wave 64x64 macro-tile GEMM inner loop: A[M][K] bf16 row-major,
// Bt[N][K] bf16 (weights pre-transposed so K is contiguous).
// 16 WMMAs per 16 b128 loads per k-step (1.0 loads/WMMA).
__device__ __forceinline__ void gemm_wave64(const __bf16* __restrict__ A,
                                            const __bf16* __restrict__ Bt, int K,
                                            int rowTile, int colTile, int lane,
                                            v8f acc[4][4]) {
    for (int k0 = 0; k0 < K; k0 += 32) {
        v16bf a[4];
#pragma unroll
        for (int i = 0; i < 4; ++i)
            a[i] = load_a_frag(A, K, rowTile + i * 16, k0, lane);
#pragma unroll
        for (int j = 0; j < 4; ++j) {
            v16bf bf = load_b_frag(Bt, K, colTile + j * 16, k0, lane);
#pragma unroll
            for (int i = 0; i < 4; ++i)
                acc[i][j] = wmma_bf16(a[i], bf, acc[i][j]);
        }
    }
}

// ---------------------------------------------------------------------------
// Prep kernels
// ---------------------------------------------------------------------------
__global__ void cvt_bf16_kernel(const float* __restrict__ x, __bf16* __restrict__ y,
                                size_t n) {
    size_t stride = (size_t)gridDim.x * blockDim.x;
    for (size_t i = (size_t)blockIdx.x * blockDim.x + threadIdx.x; i < n; i += stride)
        y[i] = (__bf16)x[i];
}

// Wt[n*K + k] = bf16(W[k*N + n])
__global__ void transpose_bf16_kernel(const float* __restrict__ W,
                                      __bf16* __restrict__ Wt, int K, int N) {
    size_t total  = (size_t)K * N;
    size_t stride = (size_t)gridDim.x * blockDim.x;
    for (size_t i = (size_t)blockIdx.x * blockDim.x + threadIdx.x; i < total; i += stride) {
        size_t n = i / K, k = i % K;
        Wt[n * K + k] = (__bf16)W[k * N + n];
    }
}

// ---------------------------------------------------------------------------
// QKV projection: C = X @ W + b, stored into attention-friendly layouts.
// Block = 4 waves (2x2), wave tile 64x64 -> block tile 128x128.
// vmode 0: out[((b*NH+h)*SEQ + pos)*HDIM + hd]   (Q and K)
// vmode 1: out[((b*NH+h)*HDIM + hd)*SEQ + pos]   (V transposed)
// ---------------------------------------------------------------------------
__global__ void __launch_bounds__(128)
proj_kernel(const __bf16* __restrict__ X, const __bf16* __restrict__ Wt,
            const float* __restrict__ bias, __bf16* __restrict__ out, int vmode) {
    const int K = DVX;
    const int nb = DVX / 128;                 // N-tiles of 128
    int bi = blockIdx.x / nb, bj = blockIdx.x % nb;
    int wave = threadIdx.x >> 5, lane = threadIdx.x & 31;
    int wm = wave & 1, wn = wave >> 1;
    int rowTile = bi * 128 + wm * 64;
    int colTile = bj * 128 + wn * 64;

    v8f acc[4][4];
#pragma unroll
    for (int i = 0; i < 4; ++i)
#pragma unroll
        for (int j = 0; j < 4; ++j) acc[i][j] = (v8f){0, 0, 0, 0, 0, 0, 0, 0};

    gemm_wave64(X, Wt, K, rowTile, colTile, lane, acc);

    int kh = lane >> 4, ln = lane & 15;
#pragma unroll
    for (int i = 0; i < 4; ++i)
#pragma unroll
        for (int j = 0; j < 4; ++j)
#pragma unroll
            for (int r = 0; r < 8; ++r) {
                int row = rowTile + i * 16 + r + 8 * kh;
                int col = colTile + j * 16 + ln;
                float v = acc[i][j][r] + bias[col];
                int b  = row >> 11;            // SEQ = 2048
                int pos = row & (SEQ - 1);
                int h  = col >> 6;             // HDIM = 64
                int hd = col & (HDIM - 1);
                size_t idx = vmode
                    ? ((size_t)((b * NH + h) * HDIM + hd) * SEQ + pos)
                    : ((size_t)((b * NH + h) * SEQ + pos) * HDIM + hd);
                out[idx] = (__bf16)v;
            }
}

// ---------------------------------------------------------------------------
// Flash attention: one block = 128 q-rows of one (b,h); 8 waves x 16 rows.
// Streaming softmax over 64-key blocks, all matmuls via v_wmma bf16.
// ---------------------------------------------------------------------------
__global__ void __launch_bounds__(256)
attn_kernel(const __bf16* __restrict__ Q, const __bf16* __restrict__ Kmat,
            const __bf16* __restrict__ Vt, const unsigned char* __restrict__ mask,
            __bf16* __restrict__ O) {
    __shared__ __align__(32) __bf16 pshm[8][16][64];  // per-wave P staging (16KB)

    const int nq = SEQ / 128;
    int qt = blockIdx.x % nq;
    int bh = blockIdx.x / nq;
    int b = bh / NH, h = bh % NH;
    int wave = threadIdx.x >> 5, lane = threadIdx.x & 31;
    int kh = lane >> 4, ln = lane & 15;

    const __bf16* Qb = Q    + (size_t)bh * SEQ * HDIM;
    const __bf16* Kb = Kmat + (size_t)bh * SEQ * HDIM;
    const __bf16* Vb = Vt   + (size_t)bh * HDIM * SEQ;
    const unsigned char* mb = mask + (size_t)b * SEQ;
    int qbase = qt * 128 + wave * 16;

    // Q fragments for this wave's 16 rows (HD=64 -> two K-steps), kept resident.
    v16bf aq[2];
    aq[0] = load_a_frag(Qb, HDIM, qbase, 0,  lane);
    aq[1] = load_a_frag(Qb, HDIM, qbase, 32, lane);

    float mrow[8], lrow[8];
    v8f oacc[4];
#pragma unroll
    for (int r = 0; r < 8; ++r) { mrow[r] = -3.0e38f; lrow[r] = 0.f; }
#pragma unroll
    for (int j = 0; j < 4; ++j) oacc[j] = (v8f){0, 0, 0, 0, 0, 0, 0, 0};

    for (int kb = 0; kb < SEQ; kb += 64) {
        // mask is monotonic (pad = tail): skip remaining blocks, uniform branch.
        if (mb[kb]) break;

        // ---- S = Q K^T (16x64 per wave) ----
        v8f s[4];
#pragma unroll
        for (int j = 0; j < 4; ++j) s[j] = (v8f){0, 0, 0, 0, 0, 0, 0, 0};
#pragma unroll
        for (int ks = 0; ks < 2; ++ks)
#pragma unroll
            for (int j = 0; j < 4; ++j) {
                v16bf bk = load_b_frag(Kb, HDIM, kb + j * 16, ks * 32, lane);
                s[j] = wmma_bf16(aq[ks], bk, s[j]);
            }

        // ---- scale (1/sqrt(64)) + key-padding mask ----
#pragma unroll
        for (int j = 0; j < 4; ++j) {
            int key = kb + j * 16 + ln;
            float pen = mb[key] ? -3.0e38f : 0.f;
#pragma unroll
            for (int r = 0; r < 8; ++r) s[j][r] = s[j][r] * 0.125f + pen;
        }

        // ---- streaming softmax stats (row lives in one 16-lane half) ----
        float mnew[8];
#pragma unroll
        for (int r = 0; r < 8; ++r) {
            float mx = fmaxf(fmaxf(s[0][r], s[1][r]), fmaxf(s[2][r], s[3][r]));
#pragma unroll
            for (int off = 1; off < 16; off <<= 1)
                mx = fmaxf(mx, __shfl_xor(mx, off, 32));
            mnew[r] = fmaxf(mrow[r], mx);
        }
#pragma unroll
        for (int r = 0; r < 8; ++r) {
            float ps = 0.f;
#pragma unroll
            for (int j = 0; j < 4; ++j) {
                float p = __expf(s[j][r] - mnew[r]);
                s[j][r] = p;
                ps += p;
            }
#pragma unroll
            for (int off = 1; off < 16; off <<= 1) ps += __shfl_xor(ps, off, 32);
            float alpha = __expf(mrow[r] - mnew[r]);
            lrow[r] = lrow[r] * alpha + ps;
            mrow[r] = mnew[r];
#pragma unroll
            for (int j = 0; j < 4; ++j) oacc[j][r] *= alpha;
        }

        // ---- repack P (C-layout) -> A-fragment layout via wave-private LDS ----
#pragma unroll
        for (int j = 0; j < 4; ++j)
#pragma unroll
            for (int r = 0; r < 8; ++r)
                pshm[wave][r + 8 * kh][j * 16 + ln] = (__bf16)s[j][r];
        asm volatile("s_wait_dscnt 0" ::: "memory");

        // ---- O += P V (V pre-transposed: rows are head-dims, contiguous keys) ----
#pragma unroll
        for (int ks = 0; ks < 2; ++ks) {
            const __bf16* pp = &pshm[wave][ln][ks * 32 + kh * 8];
            v8bf lo = *(const v8bf*)pp;
            v8bf hi = *(const v8bf*)(pp + 16);
            v16bf ap;
#pragma unroll
            for (int i = 0; i < 8; ++i) { ap[i] = lo[i]; ap[i + 8] = hi[i]; }
#pragma unroll
            for (int j = 0; j < 4; ++j) {
                v16bf bv = load_b_frag(Vb, SEQ, j * 16, kb + ks * 32, lane);
                oacc[j] = wmma_bf16(ap, bv, oacc[j]);
            }
        }
    }

    // ---- normalize and store bf16 attention output [token][DV] ----
#pragma unroll
    for (int r = 0; r < 8; ++r) {
        float inv = 1.f / lrow[r];
        int qrow = qbase + r + 8 * kh;
#pragma unroll
        for (int j = 0; j < 4; ++j) {
            int dv = h * HDIM + j * 16 + ln;
            O[(size_t)(b * SEQ + qrow) * DVX + dv] = (__bf16)(oacc[j][r] * inv);
        }
    }
}

// ---------------------------------------------------------------------------
// Generic WMMA GEMM + epilogue (4 waves, 64x64 per wave, 128x128 per block).
// mode 0: out_f32 = acc + bias + resid          (O-proj / FFN2)
// mode 1: out_bf16 = gelu_exact(acc + bias)     (FFN1)
// ---------------------------------------------------------------------------
__global__ void __launch_bounds__(128)
gemm_ep_kernel(const __bf16* __restrict__ A, const __bf16* __restrict__ Wt,
               const float* __restrict__ bias, const float* __restrict__ resid,
               float* __restrict__ out_f32, __bf16* __restrict__ out_bf16,
               int N, int K, int mode) {
    int nb = N / 128;
    int bi = blockIdx.x / nb, bj = blockIdx.x % nb;
    int wave = threadIdx.x >> 5, lane = threadIdx.x & 31;
    int wm = wave & 1, wn = wave >> 1;
    int rowTile = bi * 128 + wm * 64;
    int colTile = bj * 128 + wn * 64;

    v8f acc[4][4];
#pragma unroll
    for (int i = 0; i < 4; ++i)
#pragma unroll
        for (int j = 0; j < 4; ++j) acc[i][j] = (v8f){0, 0, 0, 0, 0, 0, 0, 0};

    gemm_wave64(A, Wt, K, rowTile, colTile, lane, acc);

    int kh = lane >> 4, ln = lane & 15;
#pragma unroll
    for (int i = 0; i < 4; ++i)
#pragma unroll
        for (int j = 0; j < 4; ++j)
#pragma unroll
            for (int r = 0; r < 8; ++r) {
                int row = rowTile + i * 16 + r + 8 * kh;
                int col = colTile + j * 16 + ln;
                size_t idx = (size_t)row * N + col;
                float v = acc[i][j][r] + bias[col];
                if (mode == 0) {
                    out_f32[idx] = v + resid[idx];
                } else {
                    float g = 0.5f * v * (1.f + erff(v * 0.70710678118654752f));
                    out_bf16[idx] = (__bf16)g;
                }
            }
}

// ---------------------------------------------------------------------------
// LayerNorm over DV=512, one block (256 threads) per token row.
// Writes f32 (always) and optionally bf16 (feeds next bf16 GEMM).
// ---------------------------------------------------------------------------
__global__ void __launch_bounds__(256)
ln_kernel(const float* __restrict__ y, const float* __restrict__ g,
          const float* __restrict__ beta, float* __restrict__ of32,
          __bf16* __restrict__ obf, int write_bf) {
    __shared__ float red[256];
    int row = blockIdx.x;
    int t = threadIdx.x;
    const float* yr = y + (size_t)row * DVX;
    float v0 = yr[t], v1 = yr[t + 256];

    red[t] = v0 + v1;
    __syncthreads();
    for (int o = 128; o > 0; o >>= 1) {
        if (t < o) red[t] += red[t + o];
        __syncthreads();
    }
    float mu = red[0] * (1.f / DVX);
    __syncthreads();

    float d0 = v0 - mu, d1 = v1 - mu;
    red[t] = d0 * d0 + d1 * d1;
    __syncthreads();
    for (int o = 128; o > 0; o >>= 1) {
        if (t < o) red[t] += red[t + o];
        __syncthreads();
    }
    float rstd = rsqrtf(red[0] * (1.f / DVX) + 1e-5f);

    float o0 = d0 * rstd * g[t] + beta[t];
    float o1 = d1 * rstd * g[t + 256] + beta[t + 256];
    of32[(size_t)row * DVX + t] = o0;
    of32[(size_t)row * DVX + t + 256] = o1;
    if (write_bf) {
        obf[(size_t)row * DVX + t] = (__bf16)o0;
        obf[(size_t)row * DVX + t + 256] = (__bf16)o1;
    }
}

// ---------------------------------------------------------------------------
// Host launcher
// ---------------------------------------------------------------------------
extern "C" void kernel_launch(void* const* d_in, const int* in_sizes, int n_in,
                              void* d_out, int out_size, void* d_ws, size_t ws_size,
                              hipStream_t stream) {
    const float* q_in = (const float*)d_in[0];
    const float* k_in = (const float*)d_in[1];
    const float* Wq   = (const float*)d_in[2];
    const float* bq   = (const float*)d_in[3];
    const float* Wk   = (const float*)d_in[4];
    const float* bk   = (const float*)d_in[5];
    const float* Wv   = (const float*)d_in[6];
    const float* bv   = (const float*)d_in[7];
    const float* Wo   = (const float*)d_in[8];
    const float* bo   = (const float*)d_in[9];
    const float* g0   = (const float*)d_in[10];
    const float* b0   = (const float*)d_in[11];
    const float* g1   = (const float*)d_in[12];
    const float* b1   = (const float*)d_in[13];
    const float* W1   = (const float*)d_in[14];
    const float* c1   = (const float*)d_in[15];
    const float* W2   = (const float*)d_in[16];
    const float* c2   = (const float*)d_in[17];
    const unsigned char* kpm = (const unsigned char*)d_in[18];

    // bump allocator over workspace (256B aligned)
    char* wp = (char*)d_ws;
    auto alloc = [&](size_t bytes) -> char* {
        char* r = wp;
        wp += (bytes + 255) & ~(size_t)255;
        return r;
    };
    const size_t actB = (size_t)BLROWS * DVX;
    __bf16* Xq  = (__bf16*)alloc(actB * 2);
    __bf16* Xk  = (__bf16*)alloc(actB * 2);
    __bf16* WqT = (__bf16*)alloc((size_t)DVX * DVX * 2);
    __bf16* WkT = (__bf16*)alloc((size_t)DVX * DVX * 2);
    __bf16* WvT = (__bf16*)alloc((size_t)DVX * DVX * 2);
    __bf16* WoT = (__bf16*)alloc((size_t)DVX * DVX * 2);
    __bf16* W1T = (__bf16*)alloc((size_t)DFF * DVX * 2);
    __bf16* W2T = (__bf16*)alloc((size_t)DVX * DFF * 2);
    __bf16* Qb  = (__bf16*)alloc(actB * 2);
    __bf16* Kb  = (__bf16*)alloc(actB * 2);
    __bf16* Vtb = (__bf16*)alloc(actB * 2);
    __bf16* AO  = (__bf16*)alloc(actB * 2);
    float*  y0  = (float*)alloc(actB * 4);
    float*  xf  = (float*)alloc(actB * 4);
    __bf16* xb  = (__bf16*)alloc(actB * 2);
    __bf16* ff1 = (__bf16*)alloc((size_t)BLROWS * DFF * 2);
    float*  y2  = y0;  // y0 dead after first LN; reuse for FFN2 output

    const dim3 blk256(256);
    const dim3 blk128(128);

    // 1) activation + weight conversion
    cvt_bf16_kernel<<<4096, blk256, 0, stream>>>(q_in, Xq, actB);
    cvt_bf16_kernel<<<4096, blk256, 0, stream>>>(k_in, Xk, actB);
    transpose_bf16_kernel<<<1024, blk256, 0, stream>>>(Wq, WqT, DVX, DVX);
    transpose_bf16_kernel<<<1024, blk256, 0, stream>>>(Wk, WkT, DVX, DVX);
    transpose_bf16_kernel<<<1024, blk256, 0, stream>>>(Wv, WvT, DVX, DVX);
    transpose_bf16_kernel<<<1024, blk256, 0, stream>>>(Wo, WoT, DVX, DVX);
    transpose_bf16_kernel<<<4096, blk256, 0, stream>>>(W1, W1T, DVX, DFF);
    transpose_bf16_kernel<<<4096, blk256, 0, stream>>>(W2, W2T, DFF, DVX);

    // 2) QKV projections (M=8192, N=512 -> 64*4 = 256 blocks of 4 waves)
    proj_kernel<<<256, blk128, 0, stream>>>(Xq, WqT, bq, Qb, 0);
    proj_kernel<<<256, blk128, 0, stream>>>(Xk, WkT, bk, Kb, 0);
    proj_kernel<<<256, blk128, 0, stream>>>(Xk, WvT, bv, Vtb, 1);

    // 3) flash attention: B*H*(SEQ/128) = 4*8*16 = 512 blocks
    attn_kernel<<<512, blk256, 0, stream>>>(Qb, Kb, Vtb, kpm, AO);

    // 4) O-projection + bias + residual(q_in)
    gemm_ep_kernel<<<256, blk128, 0, stream>>>(AO, WoT, bo, q_in, y0, nullptr,
                                               DVX, DVX, 0);
    // 5) LayerNorm 0 -> x (f32 + bf16)
    ln_kernel<<<BLROWS, blk256, 0, stream>>>(y0, g0, b0, xf, xb, 1);

    // 6) FFN1: gelu(x @ W1 + c1) -> bf16   (M=8192, N=2048 -> 1024 blocks)
    gemm_ep_kernel<<<1024, blk128, 0, stream>>>(xb, W1T, c1, nullptr, nullptr, ff1,
                                                DFF, DVX, 1);
    // 7) FFN2: ff1 @ W2 + c2 + x
    gemm_ep_kernel<<<256, blk128, 0, stream>>>(ff1, W2T, c2, xf, y2, nullptr,
                                               DVX, DFF, 0);
    // 8) LayerNorm 1 -> d_out (f32)
    ln_kernel<<<BLROWS, blk256, 0, stream>>>(y2, g1, b1, (float*)d_out, nullptr, 0);
}
